// POMMPNN_10771777978620
// MI455X (gfx1250) — matrix operand order repaired and checked by reading, not compiled, not run on hardware
//
#include <hip/hip_runtime.h>
#include <math.h>

#define N_NODES 100000
#define N_EDGES 600000
#define N_GRAPHS 2000

typedef _Float16 v16h __attribute__((ext_vector_type(16)));
typedef float    v8f  __attribute__((ext_vector_type(8)));

union H16 { float4 q[2]; v16h v; };

__device__ __forceinline__ v8f wmma16(const H16& a, const H16& b, v8f c) {
    return __builtin_amdgcn_wmma_f32_16x16x32_f16(false, a.v, false, b.v,
                                                  (short)0, c, false, false);
}

__device__ __forceinline__ void atomicMaxF(float* addr, float v) {
    if (v >= 0.0f) atomicMax((int*)addr, __float_as_int(v));
    else           atomicMin((unsigned int*)addr, __float_as_uint(v));
}

// issue one per-lane async b128 copy: global -> LDS (tracked by ASYNCcnt)
__device__ __forceinline__ void async_b128(void* lds, const void* gptr) {
    unsigned      loff = (unsigned)(uintptr_t)lds;        // low 32 bits = LDS offset
    unsigned long long ga = (unsigned long long)(uintptr_t)gptr;
    asm volatile("global_load_async_to_lds_b128 %0, %1, off"
                 :: "v"(loff), "v"(ga) : "memory");
}
__device__ __forceinline__ void async_wait0() {
    asm volatile("s_wait_asynccnt 0x0" ::: "memory");
}

// ---------------------------------------------------------------- encoder
// x = LN(relu(af @ enc_w + enc_b)) ; one wave32 per node, 4 channels/lane
__global__ __launch_bounds__(256) void mpnn_encode(
    const float* __restrict__ af, const float* __restrict__ ew,
    const float* __restrict__ eb, const float* __restrict__ lg,
    const float* __restrict__ lb, float* __restrict__ x,
    _Float16* __restrict__ xh, int n)
{
    int wave = threadIdx.x >> 5, lane = threadIdx.x & 31;
    int node = blockIdx.x * 8 + wave;
    if (node >= n) return;
    float a0 = af[node*4+0], a1 = af[node*4+1], a2 = af[node*4+2], a3 = af[node*4+3];
    float v[4]; float s = 0.0f;
    #pragma unroll
    for (int u = 0; u < 4; ++u) {
        int c = lane*4 + u;
        float t = a0*ew[c] + a1*ew[128+c] + a2*ew[256+c] + a3*ew[384+c] + eb[c];
        t = t > 0.0f ? t : 0.0f;
        v[u] = t; s += t;
    }
    #pragma unroll
    for (int m = 1; m < 32; m <<= 1) s += __shfl_xor(s, m, 32);
    float mean = s * (1.0f/128.0f);
    float vs = 0.0f;
    #pragma unroll
    for (int u = 0; u < 4; ++u) { float d = v[u]-mean; vs += d*d; }
    #pragma unroll
    for (int m = 1; m < 32; m <<= 1) vs += __shfl_xor(vs, m, 32);
    float inv = rsqrtf(vs * (1.0f/128.0f) + 1e-5f);
    size_t base = (size_t)node * 128;
    #pragma unroll
    for (int u = 0; u < 4; ++u) {
        int c = lane*4 + u;
        float o = (v[u]-mean)*inv*lg[c] + lb[c];
        x[base+c] = o; xh[base+c] = (_Float16)o;
    }
}

// ---------------------------------------------------------------- zero fill
__global__ void mpnn_zero(float* __restrict__ p, size_t n) {
    size_t i = (size_t)blockIdx.x * blockDim.x + threadIdx.x;
    size_t stride = (size_t)gridDim.x * blockDim.x;
    for (; i < n; i += stride) p[i] = 0.0f;
}

// ---------------------------------------------------------------- edge messages
// per WG: msg_w1^T [256][288] + msg_w2^T [128][256] f16 resident in LDS.
// Double-buffered A tiles filled by GLOBAL_LOAD_ASYNC_TO_LDS_B128 gathers;
// relu([16,288]@W1 + b1) @ W2 + b2 -> global_atomic_add_f32 scatter.
__global__ __launch_bounds__(256) void mpnn_edge_msg(
    const _Float16* __restrict__ xh, const float* __restrict__ eattr,
    const int* __restrict__ eidx, const float* __restrict__ w1,
    const float* __restrict__ b1, const float* __restrict__ w2,
    const float* __restrict__ b2, float* __restrict__ aggr, int E)
{
    extern __shared__ char smem_raw[];
    _Float16* Wt1 = (_Float16*)smem_raw;     // [256][288]  (col-major weights, padded K)
    _Float16* Wt2 = Wt1 + 256*288;           // [128][256]
    _Float16* At  = Wt2 + 128*256;           // 2 x [16][288]  double buffer
    _Float16* Hid = At  + 2*16*288;          // [16][256]
    __shared__ int dstS[2][16];

    const int tid = threadIdx.x;
    for (int i = tid; i < 256*288; i += 256) {
        int nn = i / 288, k = i % 288;
        Wt1[i] = (k < 257) ? (_Float16)w1[k*256 + nn] : (_Float16)0.0f;
    }
    for (int i = tid; i < 128*256; i += 256) {
        int nn = i >> 8, k = i & 255;
        Wt2[i] = (_Float16)w2[k*128 + nn];
    }
    for (int i = tid; i < 2*16*32; i += 256) {        // K padding 256..287, both buffers
        int bf = i >> 9, m = (i >> 5) & 15, k = 256 + (i & 31);
        At[bf*(16*288) + m*288 + k] = (_Float16)0.0f;
    }
    __syncthreads();

    const int wave = tid >> 5, lane = tid & 31;
    const int lm = lane & 15, lh = lane >> 4;
    const int srow = tid >> 4, part = tid & 15;       // staging role: 16B chunk per thread

    const int ntiles = E >> 4;
    const int stride = gridDim.x;

    // async-gather one 16-edge tile into buffer bf
    auto issue = [&](int tile, int bf) {
        int e = (tile << 4) + srow;
        int s = eidx[e];
        int d = eidx[E + e];
        _Float16* Ab = At + bf*(16*288) + srow*288;
        async_b128(Ab + part*8,        xh + (size_t)s*128 + part*8);
        async_b128(Ab + 128 + part*8,  xh + (size_t)d*128 + part*8);
        if (part == 0) Ab[256] = (_Float16)eattr[e];
        if (part == 1) dstS[bf][srow] = d;
    };

    int tile = blockIdx.x;
    int buf = 0;
    if (tile < ntiles) issue(tile, buf);

    for (; tile < ntiles; tile += stride) {
        const int cur = buf;
        async_wait0();            // my gather for `cur` landed in LDS
        __syncthreads();          // everyone's gather landed

        int nxt = tile + stride;
        buf ^= 1;
        if (nxt < ntiles) issue(nxt, buf);   // prefetch next tile while we compute

        const _Float16* Ac = At + cur*(16*288);

        // hoist A fragments: reused by both N-tiles of GEMM1
        H16 aR[9];
        #pragma unroll
        for (int kk = 0; kk < 9; ++kk) {
            aR[kk].q[0] = *(const float4*)(Ac + lm*288 + kk*32 + lh*8);
            aR[kk].q[1] = *(const float4*)(Ac + lm*288 + kk*32 + 16 + lh*8);
        }
        // GEMM1: [16,288] x [288,256], relu -> Hid (f16)
        #pragma unroll
        for (int i = 0; i < 2; ++i) {
            int col = (wave*2 + i)*16 + lm;
            v8f acc = {};
            #pragma unroll
            for (int kk = 0; kk < 9; ++kk) {
                int k0 = kk*32;
                H16 b;
                b.q[0] = *(const float4*)(Wt1 + col*288 + k0 + lh*8);
                b.q[1] = *(const float4*)(Wt1 + col*288 + k0 + 16 + lh*8);
                acc = wmma16(aR[kk], b, acc);
            }
            float bias = b1[col];
            #pragma unroll
            for (int j = 0; j < 8; ++j) {
                float v = acc[j] + bias;
                v = v > 0.0f ? v : 0.0f;
                Hid[(j + lh*8)*256 + col] = (_Float16)v;
            }
        }
        __syncthreads();
        // GEMM2: [16,256] x [256,128] + b2 -> scatter-add into aggr[dst]
        {
            int col = wave*16 + lm;
            v8f acc = {};
            #pragma unroll
            for (int kk = 0; kk < 8; ++kk) {
                int k0 = kk*32;
                H16 a, b;
                a.q[0] = *(const float4*)(Hid + lm*256 + k0 + lh*8);
                a.q[1] = *(const float4*)(Hid + lm*256 + k0 + 16 + lh*8);
                b.q[0] = *(const float4*)(Wt2 + col*256 + k0 + lh*8);
                b.q[1] = *(const float4*)(Wt2 + col*256 + k0 + 16 + lh*8);
                acc = wmma16(a, b, acc);
            }
            float bias = b2[col];
            #pragma unroll
            for (int j = 0; j < 8; ++j) {
                int d = dstS[cur][j + lh*8];
                atomicAdd(aggr + (size_t)d*128 + col, acc[j] + bias);
            }
        }
        __syncthreads();   // Hid (and old A buffer) free for reuse
    }
}

// ---------------------------------------------------------------- GRU + LayerNorm
__global__ __launch_bounds__(256) void mpnn_gru(
    const float* __restrict__ aggr, float* __restrict__ x,
    _Float16* __restrict__ xh, const float* __restrict__ wih,
    const float* __restrict__ whh, const float* __restrict__ bih,
    const float* __restrict__ bhh, const float* __restrict__ lng,
    const float* __restrict__ lnb, int n)
{
    extern __shared__ char smem_raw[];
    _Float16* Wih = (_Float16*)smem_raw;      // [384][128]  (row j = output col, contiguous k)
    _Float16* Whh = Wih + 384*128;
    _Float16* Ag  = Whh + 384*128;            // [16][128]
    _Float16* Xt  = Ag  + 16*128;
    float*    Gi  = (float*)(Xt + 16*128);    // [16][384]
    float*    Gh  = Gi + 16*384;

    const int tid = threadIdx.x;
    for (int i = tid; i < 384*128; i += 256) {
        Wih[i] = (_Float16)wih[i];
        Whh[i] = (_Float16)whh[i];
    }
    __syncthreads();

    const int wave = tid >> 5, lane = tid & 31;
    const int lm = lane & 15, lh = lane >> 4;

    int ntiles = n >> 4;
    for (int tile = blockIdx.x; tile < ntiles; tile += gridDim.x) {
        int n0 = tile << 4;
        for (int i = tid; i < 16*128; i += 256) {
            int r = i >> 7, c = i & 127;
            size_t g = (size_t)(n0 + r)*128 + c;
            Ag[i] = (_Float16)aggr[g];
            Xt[i] = (_Float16)x[g];
        }
        __syncthreads();
        // hoist A fragments: reused across 3 column tiles x 2 GEMMs
        H16 aA[4], aX[4];
        #pragma unroll
        for (int kk = 0; kk < 4; ++kk) {
            aA[kk].q[0] = *(const float4*)(Ag + lm*128 + kk*32 + lh*8);
            aA[kk].q[1] = *(const float4*)(Ag + lm*128 + kk*32 + 16 + lh*8);
            aX[kk].q[0] = *(const float4*)(Xt + lm*128 + kk*32 + lh*8);
            aX[kk].q[1] = *(const float4*)(Xt + lm*128 + kk*32 + 16 + lh*8);
        }
        #pragma unroll
        for (int i = 0; i < 3; ++i) {
            int col = (wave + i*8)*16 + lm;
            v8f ai = {}, ah = {};
            #pragma unroll
            for (int kk = 0; kk < 4; ++kk) {
                int k0 = kk*32;
                H16 b;
                b.q[0] = *(const float4*)(Wih + col*128 + k0 + lh*8);
                b.q[1] = *(const float4*)(Wih + col*128 + k0 + 16 + lh*8);
                ai = wmma16(aA[kk], b, ai);
                b.q[0] = *(const float4*)(Whh + col*128 + k0 + lh*8);
                b.q[1] = *(const float4*)(Whh + col*128 + k0 + 16 + lh*8);
                ah = wmma16(aX[kk], b, ah);
            }
            float bi = bih[col], bh = bhh[col];
            #pragma unroll
            for (int j = 0; j < 8; ++j) {
                int r = j + lh*8;
                Gi[r*384 + col] = ai[j] + bi;
                Gh[r*384 + col] = ah[j] + bh;
            }
        }
        __syncthreads();
        {   // gates + residual + LayerNorm; 16 lanes per row, 8 cols per lane
            int r = tid >> 4, seg = tid & 15;
            size_t nrow = (size_t)(n0 + r) * 128;
            float pre[8]; float s = 0.0f;
            #pragma unroll
            for (int u = 0; u < 8; ++u) {
                int c = seg*8 + u;
                float ir = Gi[r*384 + c], iz = Gi[r*384 + 128 + c], inn = Gi[r*384 + 256 + c];
                float hr = Gh[r*384 + c], hz = Gh[r*384 + 128 + c], hnn = Gh[r*384 + 256 + c];
                float rr = 1.0f/(1.0f + expf(-(ir + hr)));
                float zz = 1.0f/(1.0f + expf(-(iz + hz)));
                float nnv = tanhf(inn + rr*hnn);
                float xv = x[nrow + c];
                float xn = (1.0f - zz)*nnv + zz*xv;
                pre[u] = xv + xn;
                s += pre[u];
            }
            #pragma unroll
            for (int m = 1; m < 16; m <<= 1) s += __shfl_xor(s, m, 32);
            float mean = s * (1.0f/128.0f);
            float vs = 0.0f;
            #pragma unroll
            for (int u = 0; u < 8; ++u) { float d = pre[u]-mean; vs += d*d; }
            #pragma unroll
            for (int m = 1; m < 16; m <<= 1) vs += __shfl_xor(vs, m, 32);
            float inv = rsqrtf(vs * (1.0f/128.0f) + 1e-5f);
            #pragma unroll
            for (int u = 0; u < 8; ++u) {
                int c = seg*8 + u;
                float o = (pre[u]-mean)*inv*lng[c] + lnb[c];
                x[nrow + c] = o; xh[nrow + c] = (_Float16)o;
            }
        }
        __syncthreads();
    }
}

// ---------------------------------------------------------------- pooling
__global__ void mpnn_pool_init(float* __restrict__ sums, float* __restrict__ maxs,
                               float* __restrict__ counts, int G)
{
    int i = blockIdx.x * blockDim.x + threadIdx.x;
    if (i < G*128) { sums[i] = 0.0f; maxs[i] = -3.402823466e38f; }
    if (i < G) counts[i] = 0.0f;
}

__global__ void mpnn_pool(const float* __restrict__ x, const int* __restrict__ batch,
                          float* __restrict__ sums, float* __restrict__ maxs,
                          float* __restrict__ counts, int n)
{
    size_t i = (size_t)blockIdx.x * blockDim.x + threadIdx.x;
    size_t stride = (size_t)gridDim.x * blockDim.x;
    size_t total = (size_t)n * 128;
    for (; i < total; i += stride) {
        int node = (int)(i >> 7), c = (int)(i & 127);
        int g = batch[node];
        float v = x[i];
        atomicAdd(&sums[g*128 + c], v);
        atomicMaxF(&maxs[g*128 + c], v);
        if (c == 0) atomicAdd(&counts[g], 1.0f);
    }
}

__global__ void mpnn_pool_final(const float* __restrict__ sums, const float* __restrict__ maxs,
                                const float* __restrict__ counts, float* __restrict__ gbuf, int G)
{
    int i = blockIdx.x * blockDim.x + threadIdx.x;
    if (i >= G*256) return;
    int g = i >> 8, c = i & 255;
    gbuf[i] = (c < 128) ? sums[g*128 + c] / fmaxf(counts[g], 1.0f)
                        : maxs[g*128 + (c-128)];
}

// ---------------------------------------------------------------- readout MLP
__global__ void mpnn_ro(const float* __restrict__ in, const float* __restrict__ w,
                        const float* __restrict__ b, float* __restrict__ out,
                        int G, int K, int NO, int relu)
{
    int i = blockIdx.x * blockDim.x + threadIdx.x;
    if (i >= G*NO) return;
    int g = i / NO, j = i % NO;
    float s = b[j];
    const float* row = in + (size_t)g * K;
    for (int k = 0; k < K; ++k) s += row[k] * w[k*NO + j];
    out[i] = relu ? (s > 0.0f ? s : 0.0f) : s;
}

// ---------------------------------------------------------------- launch
extern "C" void kernel_launch(void* const* d_in, const int* in_sizes, int n_in,
                              void* d_out, int out_size, void* d_ws, size_t ws_size,
                              hipStream_t stream)
{
    const float* af      = (const float*)d_in[0];
    const int*   eidx    = (const int*)  d_in[1];
    const float* eattr   = (const float*)d_in[2];
    const int*   batch   = (const int*)  d_in[3];
    const float* enc_w   = (const float*)d_in[4];
    const float* enc_b   = (const float*)d_in[5];
    const float* enc_lg  = (const float*)d_in[6];
    const float* enc_lb  = (const float*)d_in[7];
    const float* msg_w1  = (const float*)d_in[8];
    const float* msg_b1  = (const float*)d_in[9];
    const float* msg_w2  = (const float*)d_in[10];
    const float* msg_b2  = (const float*)d_in[11];
    const float* gru_wih = (const float*)d_in[12];
    const float* gru_whh = (const float*)d_in[13];
    const float* gru_bih = (const float*)d_in[14];
    const float* gru_bhh = (const float*)d_in[15];
    const float* ln_g    = (const float*)d_in[16];
    const float* ln_b    = (const float*)d_in[17];
    const float* ro_w1   = (const float*)d_in[18];
    const float* ro_b1   = (const float*)d_in[19];
    const float* ro_w2   = (const float*)d_in[20];
    const float* ro_b2   = (const float*)d_in[21];
    const float* ro_w3   = (const float*)d_in[22];
    const float* ro_b3   = (const float*)d_in[23];
    (void)in_sizes; (void)n_in; (void)out_size; (void)ws_size;

    // workspace layout
    char* ws = (char*)d_ws;
    size_t off = 0;
    auto carve = [&](size_t bytes) { char* p = ws + off; off = (off + bytes + 255) & ~(size_t)255; return p; };
    float*    x      = (float*)   carve((size_t)N_NODES * 128 * 4);
    _Float16* xh     = (_Float16*)carve((size_t)N_NODES * 128 * 2);
    float*    aggr   = (float*)   carve((size_t)N_NODES * 128 * 4);
    float*    sums   = (float*)   carve((size_t)N_GRAPHS * 128 * 4);
    float*    maxs   = (float*)   carve((size_t)N_GRAPHS * 128 * 4);
    float*    counts = (float*)   carve((size_t)N_GRAPHS * 4);
    float*    gbuf   = (float*)   carve((size_t)N_GRAPHS * 256 * 4);
    float*    h1     = (float*)   carve((size_t)N_GRAPHS * 128 * 4);
    float*    h2     = (float*)   carve((size_t)N_GRAPHS * 64 * 4);

    const size_t EDGE_LDS = (size_t)(256*288 + 128*256 + 2*16*288 + 16*256) * 2; // 239616 B
    const size_t GRU_LDS  = (size_t)(384*128*2 + 16*128*2) * 2 + (size_t)16*384*4*2; // 253952 B
    (void)hipFuncSetAttribute((const void*)mpnn_edge_msg,
        hipFuncAttributeMaxDynamicSharedMemorySize, (int)EDGE_LDS);
    (void)hipFuncSetAttribute((const void*)mpnn_gru,
        hipFuncAttributeMaxDynamicSharedMemorySize, (int)GRU_LDS);

    // 1) encoder
    mpnn_encode<<<(N_NODES + 7) / 8, 256, 0, stream>>>(
        af, enc_w, enc_b, enc_lg, enc_lb, x, xh, N_NODES);

    // 2) message-passing layers
    for (int l = 0; l < 3; ++l) {
        mpnn_zero<<<2048, 256, 0, stream>>>(aggr, (size_t)N_NODES * 128);
        mpnn_edge_msg<<<2048, 256, EDGE_LDS, stream>>>(
            xh, eattr, eidx,
            msg_w1 + (size_t)l*257*256, msg_b1 + (size_t)l*256,
            msg_w2 + (size_t)l*256*128, msg_b2 + (size_t)l*128,
            aggr, N_EDGES);
        mpnn_gru<<<2048, 256, GRU_LDS, stream>>>(
            aggr, x, xh,
            gru_wih + (size_t)l*384*128, gru_whh + (size_t)l*384*128,
            gru_bih + (size_t)l*384, gru_bhh + (size_t)l*384,
            ln_g + (size_t)l*128, ln_b + (size_t)l*128, N_NODES);
    }

    // 3) pooling
    mpnn_pool_init<<<(N_GRAPHS * 128 + 255) / 256, 256, 0, stream>>>(sums, maxs, counts, N_GRAPHS);
    mpnn_pool<<<4096, 256, 0, stream>>>(x, batch, sums, maxs, counts, N_NODES);
    mpnn_pool_final<<<(N_GRAPHS * 256 + 255) / 256, 256, 0, stream>>>(sums, maxs, counts, gbuf, N_GRAPHS);

    // 4) readout MLP
    mpnn_ro<<<(N_GRAPHS * 128 + 255) / 256, 256, 0, stream>>>(gbuf, ro_w1, ro_b1, h1, N_GRAPHS, 256, 128, 1);
    mpnn_ro<<<(N_GRAPHS * 64  + 255) / 256, 256, 0, stream>>>(h1,   ro_w2, ro_b2, h2, N_GRAPHS, 128, 64, 1);
    mpnn_ro<<<(N_GRAPHS * 113 + 255) / 256, 256, 0, stream>>>(h2,   ro_w3, ro_b3, (float*)d_out, N_GRAPHS, 64, 113, 0);
}
